// MultiHeadAttention_42691974922420
// MI455X (gfx1250) — compile-verified, matrix operand order checked by reading, not run on hardware
//
#include <hip/hip_runtime.h>

// MHA forward with RoPE for MI455X (gfx1250, wave32, WMMA).
// All matmuls use v_wmma_f32_16x16x32_f16 (f16 A/B, f32 accum).
// Wave tile 64x32 (8 WMMAs per 32-wide k-step, 6 fragment loads),
// block = 8 waves (4 row x 2 col) -> 256x64 block tile.

typedef __attribute__((ext_vector_type(16))) _Float16 v16h;
typedef __attribute__((ext_vector_type(8)))  float    v8f;

#define BB 4
#define TT 1024
#define DD 1024
#define HH 16
#define HDIM 64

// ---------------- WMMA helpers (layouts per CDNA5 ISA 7.12.2) ----------------

static __device__ __forceinline__ v8f wmma16(v16h a, v16h b, v8f c) {
  return __builtin_amdgcn_wmma_f32_16x16x32_f16(
      /*neg_a=*/false, a, /*neg_b=*/false, b,
      /*c_mod=*/(short)0, c, /*reuse_a=*/false, /*reuse_b=*/false);
}

// A fragment (16x32 f16, row-major source). lanes 0-15: rows 0-15, K {k0..k0+7, k0+16..k0+23};
// lanes 16-31: same rows, K {k0+8.., k0+24..}. Two 16B contiguous loads per lane.
static __device__ __forceinline__ v16h ld_frag_h(const _Float16* __restrict__ base,
                                                 int ld, int r0, int k0) {
  const int lane = threadIdx.x & 31;
  const int row  = r0 + (lane & 15);
  const int kb   = k0 + ((lane >> 4) << 3);
  const _Float16* p = base + (size_t)row * ld + kb;
  v16h f;
  reinterpret_cast<uint4*>(&f)[0] = *reinterpret_cast<const uint4*>(p);
  reinterpret_cast<uint4*>(&f)[1] = *reinterpret_cast<const uint4*>(p + 16);
  return f;
}

// Same fragment pattern but source is f32; convert to f16 at load (softmax weights as A).
static __device__ __forceinline__ v16h ld_frag_f32(const float* __restrict__ base,
                                                   int ld, int r0, int k0) {
  const int lane = threadIdx.x & 31;
  const int row  = r0 + (lane & 15);
  const int kb   = k0 + ((lane >> 4) << 3);
  const float* p = base + (size_t)row * ld + kb;
  float4 x0 = *reinterpret_cast<const float4*>(p);
  float4 x1 = *reinterpret_cast<const float4*>(p + 4);
  float4 y0 = *reinterpret_cast<const float4*>(p + 16);
  float4 y1 = *reinterpret_cast<const float4*>(p + 20);
  v16h f;
  f[0]=(_Float16)x0.x; f[1]=(_Float16)x0.y; f[2]=(_Float16)x0.z; f[3]=(_Float16)x0.w;
  f[4]=(_Float16)x1.x; f[5]=(_Float16)x1.y; f[6]=(_Float16)x1.z; f[7]=(_Float16)x1.w;
  f[8]=(_Float16)y0.x; f[9]=(_Float16)y0.y; f[10]=(_Float16)y0.z; f[11]=(_Float16)y0.w;
  f[12]=(_Float16)y1.x; f[13]=(_Float16)y1.y; f[14]=(_Float16)y1.z; f[15]=(_Float16)y1.w;
  return f;
}

// D tile store: vgpr r holds M=r (lanes 0-15) / M=8+r (lanes 16-31), N = lane&15.
static __device__ __forceinline__ void st_tile_f16(_Float16* __restrict__ base, int ld,
                                                   int r0, int c0, const v8f& acc) {
  const int lane = threadIdx.x & 31;
  const int n  = lane & 15;
  const int mh = (lane >> 4) << 3;
#pragma unroll
  for (int r = 0; r < 8; ++r)
    base[(size_t)(r0 + mh + r) * ld + c0 + n] = (_Float16)acc[r];
}

static __device__ __forceinline__ void st_tile_f32(float* __restrict__ base, int ld,
                                                   int r0, int c0, const v8f& acc, float scale) {
  const int lane = threadIdx.x & 31;
  const int n  = lane & 15;
  const int mh = (lane >> 4) << 3;
#pragma unroll
  for (int r = 0; r < 8; ++r)
    base[(size_t)(r0 + mh + r) * ld + c0 + n] = acc[r] * scale;
}

// ---------------- generic GEMM: C[M,N] = A[M,K] * Bt[N,K]^T ----------------
// grid = (N/64, M/256). Wave tile 64x32; block tile 256x64.

template <bool OUT_F32>
__global__ __launch_bounds__(256)
void gemm_wmma_kernel(const _Float16* __restrict__ A, const _Float16* __restrict__ Bt,
                      void* __restrict__ C, int M, int N, int K) {
  const int w  = threadIdx.x >> 5;
  const int wr = w >> 1, wc = w & 1;
  const int row0 = blockIdx.y * 256 + wr * 64;
  const int col0 = blockIdx.x * 64 + wc * 32;
  v8f acc[4][2] = {};
  for (int kk = 0; kk < K; kk += 32) {
    __builtin_prefetch(A + (size_t)row0 * K + kk + 64, 0, 0);  // global_prefetch_b8
    v16h b0 = ld_frag_h(Bt, K, col0, kk);
    v16h b1 = ld_frag_h(Bt, K, col0 + 16, kk);
    v16h a[4];
#pragma unroll
    for (int i = 0; i < 4; ++i) a[i] = ld_frag_h(A, K, row0 + 16 * i, kk);
#pragma unroll
    for (int i = 0; i < 4; ++i) {
      acc[i][0] = wmma16(a[i], b0, acc[i][0]);
      acc[i][1] = wmma16(a[i], b1, acc[i][1]);
    }
  }
  if constexpr (OUT_F32) {
    float* Cf = (float*)C;
#pragma unroll
    for (int i = 0; i < 4; ++i) {
      st_tile_f32(Cf, N, row0 + 16 * i, col0,      acc[i][0], 1.0f);
      st_tile_f32(Cf, N, row0 + 16 * i, col0 + 16, acc[i][1], 1.0f);
    }
  } else {
    _Float16* Ch = (_Float16*)C;
#pragma unroll
    for (int i = 0; i < 4; ++i) {
      st_tile_f16(Ch, N, row0 + 16 * i, col0,      acc[i][0]);
      st_tile_f16(Ch, N, row0 + 16 * i, col0 + 16, acc[i][1]);
    }
  }
}

// ---------------- QK^T: scores[b,h,q,k] = scale * Qh . Kh ----------------
// grid = (T/64, T/256, B*H). Wave tile 64(q) x 32(k).

__global__ __launch_bounds__(256)
void qk_kernel(const _Float16* __restrict__ Qp, const _Float16* __restrict__ Kp,
               float* __restrict__ wout) {
  const int bh = blockIdx.z, b = bh >> 4, h = bh & 15;
  const int w  = threadIdx.x >> 5;
  const int wr = w >> 1, wc = w & 1;
  const int q0 = blockIdx.y * 256 + wr * 64;
  const int k0 = blockIdx.x * 64 + wc * 32;
  const _Float16* Qb = Qp + (size_t)b * TT * DD + h * HDIM;
  const _Float16* Kb = Kp + (size_t)b * TT * DD + h * HDIM;
  v8f acc[4][2] = {};
#pragma unroll
  for (int kk = 0; kk < HDIM; kk += 32) {
    v16h b0 = ld_frag_h(Kb, DD, k0, kk);
    v16h b1 = ld_frag_h(Kb, DD, k0 + 16, kk);
    v16h a[4];
#pragma unroll
    for (int i = 0; i < 4; ++i) a[i] = ld_frag_h(Qb, DD, q0 + 16 * i, kk);
#pragma unroll
    for (int i = 0; i < 4; ++i) {
      acc[i][0] = wmma16(a[i], b0, acc[i][0]);
      acc[i][1] = wmma16(a[i], b1, acc[i][1]);
    }
  }
  const float scale = 0.125f;  // 1/sqrt(64)
  float* Wb = wout + (size_t)bh * TT * TT;
#pragma unroll
  for (int i = 0; i < 4; ++i) {
    st_tile_f32(Wb, TT, q0 + 16 * i, k0,      acc[i][0], scale);
    st_tile_f32(Wb, TT, q0 + 16 * i, k0 + 16, acc[i][1], scale);
  }
}

// ---------------- softmax in place over last dim (row length 1024) ----------------

__global__ __launch_bounds__(256)
void softmax_kernel(float* __restrict__ wout) {
  __shared__ float red[256];
  float* row = wout + (size_t)blockIdx.x * TT;
  const int t = threadIdx.x;
  float v[4];
  float m = -3.402823466e38f;
#pragma unroll
  for (int i = 0; i < 4; ++i) { v[i] = row[t + i * 256]; m = fmaxf(m, v[i]); }
  red[t] = m; __syncthreads();
  for (int s = 128; s > 0; s >>= 1) {
    if (t < s) red[t] = fmaxf(red[t], red[t + s]);
    __syncthreads();
  }
  m = red[0]; __syncthreads();
  float sum = 0.0f;
#pragma unroll
  for (int i = 0; i < 4; ++i) { v[i] = __expf(v[i] - m); sum += v[i]; }
  red[t] = sum; __syncthreads();
  for (int s = 128; s > 0; s >>= 1) {
    if (t < s) red[t] += red[t + s];
    __syncthreads();
  }
  const float inv = 1.0f / red[0];
#pragma unroll
  for (int i = 0; i < 4; ++i) row[t + i * 256] = v[i] * inv;
}

// ---------------- AV: Y[b,h,t,d] = sum_k w[b,h,t,k] * Vh[k,d] ----------------
// VpT is (bh, d, t) so B-fragments are contiguous. grid = (1, T/256, B*H).
// Wave tile 64(t) x 32(d); block tile 256x64 (full head dim).

__global__ __launch_bounds__(256)
void av_kernel(const float* __restrict__ wsm, const _Float16* __restrict__ VpT,
               _Float16* __restrict__ Y) {
  const int bh = blockIdx.z, b = bh >> 4, h = bh & 15;
  const int w  = threadIdx.x >> 5;
  const int wr = w >> 1, wc = w & 1;
  const int t0 = blockIdx.y * 256 + wr * 64;
  const int d0 = wc * 32;
  const float* Wb = wsm + (size_t)bh * TT * TT;
  const _Float16* Vb = VpT + (size_t)bh * HDIM * TT;
  v8f acc[4][2] = {};
  for (int kk = 0; kk < TT; kk += 32) {
    __builtin_prefetch(Wb + (size_t)t0 * TT + kk + 32, 0, 0);
    v16h b0 = ld_frag_h(Vb, TT, d0, kk);
    v16h b1 = ld_frag_h(Vb, TT, d0 + 16, kk);
    v16h a[4];
#pragma unroll
    for (int i = 0; i < 4; ++i) a[i] = ld_frag_f32(Wb, TT, t0 + 16 * i, kk);
#pragma unroll
    for (int i = 0; i < 4; ++i) {
      acc[i][0] = wmma16(a[i], b0, acc[i][0]);
      acc[i][1] = wmma16(a[i], b1, acc[i][1]);
    }
  }
  _Float16* Yb = Y + (size_t)b * TT * DD + h * HDIM;
#pragma unroll
  for (int i = 0; i < 4; ++i) {
    st_tile_f16(Yb, DD, t0 + 16 * i, d0,      acc[i][0]);
    st_tile_f16(Yb, DD, t0 + 16 * i, d0 + 16, acc[i][1]);
  }
}

// ---------------- small elementwise kernels ----------------

__global__ __launch_bounds__(256)
void cvt_f32_f16_kernel(const float* __restrict__ in, _Float16* __restrict__ out, int n) {
  int i = blockIdx.x * 256 + threadIdx.x;
  if (i < n) out[i] = (_Float16)in[i];
}

// out[n,k] = (f16) in[k,n]  (weights: store transposed for B-fragment loads)
__global__ __launch_bounds__(256)
void cvt_transpose_kernel(const float* __restrict__ in, _Float16* __restrict__ out) {
  int i = blockIdx.x * 256 + threadIdx.x;
  if (i >= DD * DD) return;
  int n = i / DD, k = i - n * DD;
  out[i] = (_Float16)in[(size_t)k * DD + n];
}

// In-place RoPE on projected Q or K (rows = B*T, cols = D, pairs within each 64-wide head)
__global__ __launch_bounds__(256)
void rope_kernel(_Float16* __restrict__ X) {
  int i = blockIdx.x * 256 + threadIdx.x;
  if (i >= BB * TT * (DD / 2)) return;
  int row  = i >> 9;          // / (D/2)
  int pair = i & 511;
  int c = pair * 2;
  int d = c & (HDIM - 1);     // even dim index within head
  int t = row & (TT - 1);
  float ang = (float)t * powf(10000.0f, -(float)d / (float)HDIM);
  float sn, cs;
  sincosf(ang, &sn, &cs);
  _Float16* p = X + (size_t)row * DD + c;
  float u1 = (float)p[0], u2 = (float)p[1];
  p[0] = (_Float16)(u1 * cs - u2 * sn);
  p[1] = (_Float16)(u2 * cs + u1 * sn);
}

// VpT[(b*H+h)*HD + d][t] = Vp[b*T+t][h*HD+d]
__global__ __launch_bounds__(256)
void vtrans_kernel(const _Float16* __restrict__ Vp, _Float16* __restrict__ VpT) {
  int i = blockIdx.x * 256 + threadIdx.x;
  if (i >= BB * TT * DD) return;
  int col = i & (DD - 1);
  int row = i >> 10;
  int b = row >> 10, t = row & (TT - 1);
  int h = col >> 6, d = col & (HDIM - 1);
  VpT[((size_t)(b * HH + h) * HDIM + d) * TT + t] = Vp[i];
}

// ---------------- host launcher ----------------

extern "C" void kernel_launch(void* const* d_in, const int* in_sizes, int n_in,
                              void* d_out, int out_size, void* d_ws, size_t ws_size,
                              hipStream_t stream) {
  (void)in_sizes; (void)n_in; (void)out_size; (void)ws_size;
  const float* q  = (const float*)d_in[0];
  const float* k  = (const float*)d_in[1];
  const float* v  = (const float*)d_in[2];
  const float* Wq = (const float*)d_in[3];
  const float* Wk = (const float*)d_in[4];
  const float* Wv = (const float*)d_in[5];
  const float* Wo = (const float*)d_in[6];

  float* logits = (float*)d_out;                          // B*T*D
  float* wout   = logits + (size_t)BB * TT * DD;          // B*H*T*T

  char* wsb = (char*)d_ws;
  const size_t MB = 1ull << 20;
  _Float16* qh  = (_Float16*)(wsb + 0 * MB);    // 8 MB each for activations
  _Float16* kh  = (_Float16*)(wsb + 8 * MB);
  _Float16* vh  = (_Float16*)(wsb + 16 * MB);
  _Float16* WqT = (_Float16*)(wsb + 24 * MB);   // 2 MB each
  _Float16* WkT = (_Float16*)(wsb + 26 * MB);
  _Float16* WvT = (_Float16*)(wsb + 28 * MB);
  _Float16* WoT = (_Float16*)(wsb + 30 * MB);
  _Float16* Qp  = (_Float16*)(wsb + 32 * MB);
  _Float16* Kp  = (_Float16*)(wsb + 40 * MB);
  _Float16* Vp  = (_Float16*)(wsb + 48 * MB);
  _Float16* VpT = (_Float16*)(wsb + 56 * MB);
  _Float16* Yh  = (_Float16*)(wsb + 64 * MB);   // total 72 MB

  const int nAct = BB * TT * DD;                // 4M
  const int nW   = DD * DD;                     // 1M

  // 1) convert activations f32 -> f16
  cvt_f32_f16_kernel<<<(nAct + 255) / 256, 256, 0, stream>>>(q, qh, nAct);
  cvt_f32_f16_kernel<<<(nAct + 255) / 256, 256, 0, stream>>>(k, kh, nAct);
  cvt_f32_f16_kernel<<<(nAct + 255) / 256, 256, 0, stream>>>(v, vh, nAct);

  // 2) transpose + convert weights
  cvt_transpose_kernel<<<(nW + 255) / 256, 256, 0, stream>>>(Wq, WqT);
  cvt_transpose_kernel<<<(nW + 255) / 256, 256, 0, stream>>>(Wk, WkT);
  cvt_transpose_kernel<<<(nW + 255) / 256, 256, 0, stream>>>(Wv, WvT);
  cvt_transpose_kernel<<<(nW + 255) / 256, 256, 0, stream>>>(Wo, WoT);

  // 3) Q/K/V projections (WMMA)
  dim3 gp(DD / 64, (BB * TT) / 256, 1);
  gemm_wmma_kernel<false><<<gp, 256, 0, stream>>>(qh, WqT, Qp, BB * TT, DD, DD);
  gemm_wmma_kernel<false><<<gp, 256, 0, stream>>>(kh, WkT, Kp, BB * TT, DD, DD);
  gemm_wmma_kernel<false><<<gp, 256, 0, stream>>>(vh, WvT, Vp, BB * TT, DD, DD);

  // 4) RoPE on Q,K; transpose V per head
  int nPair = BB * TT * (DD / 2);
  rope_kernel<<<(nPair + 255) / 256, 256, 0, stream>>>(Qp);
  rope_kernel<<<(nPair + 255) / 256, 256, 0, stream>>>(Kp);
  vtrans_kernel<<<(nAct + 255) / 256, 256, 0, stream>>>(Vp, VpT);

  // 5) scores = scale * Q K^T  -> w output region (pre-softmax)
  dim3 gqk(TT / 64, TT / 256, BB * HH);
  qk_kernel<<<gqk, 256, 0, stream>>>(Qp, Kp, wout);

  // 6) softmax in place
  softmax_kernel<<<BB * HH * TT, 256, 0, stream>>>(wout);

  // 7) Y = w @ V   (A fragments converted f32->f16 at load)
  dim3 gav(1, TT / 256, BB * HH);
  av_kernel<<<gav, 256, 0, stream>>>(wout, VpT, Yh);

  // 8) logits = Y @ Wo
  gemm_wmma_kernel<true><<<gp, 256, 0, stream>>>(Yh, WoT, logits, BB * TT, DD, DD);
}
